// TSClusteringLayer_70145405878890
// MI455X (gfx1250) — matrix operand order repaired and checked by reading, not compiled, not run on hardware
//
#include <hip/hip_runtime.h>
#include <math.h>

typedef __attribute__((ext_vector_type(16))) _Float16     v16h;
typedef __attribute__((ext_vector_type(8)))  float        v8f;
typedef __attribute__((ext_vector_type(4)))  float        v4f;
typedef __attribute__((ext_vector_type(4)))  unsigned int v4u;

// x:        (N, 128, 32) f32, row-major  -> x[n*4096 + t*32 + f]
// clusters: (32, 128, 32) f32            -> c[k*4096 + t*32 + f]
// out q:    (N, 32) f32
//
// Block: 512 threads (16 wave32), one 16-row n-tile. 8 chunks = (t-chunk 0..3) x
// (f-half 0..1). Software-pipelined global->LDS staging; each wave owns exactly
// 1 feature per f-half -> only 4 x v8f WMMA accumulators per thread (32 VGPRs),
// keeping total pressure well under 256 so nothing spills to scratch.
__global__ __launch_bounds__(512)
void tscluster_wmma_kernel(const float* __restrict__ xg,
                           const float* __restrict__ cg,
                           float* __restrict__ outg)
{
    __shared__ alignas(16) _Float16 ldsX[16][16][32]; // [f'][n][t]   16 KB
    __shared__ alignas(16) _Float16 ldsC[16][32][32]; // [f'][k][t]   32 KB
    __shared__ float x2s[16][32];                     // [n][f]        2 KB
    __shared__ float c2s[32][32];                     // [k][f]        4 KB
    __shared__ float dist[16][32];                    // [n][k]        2 KB

    const int tid = threadIdx.x;
    const int w   = tid >> 5;   // wave id 0..15
    const int L   = tid & 31;   // lane
    const int g   = L >> 4;     // lane half-group
    const int m   = L & 15;     // lane within half-group
    const int n0  = blockIdx.x * 16;

    // zero accumulating LDS (512 threads)
    {
        x2s [tid >> 5][tid & 31]  = 0.0f;   // 512 entries, 1 each
        dist[tid >> 5][tid & 31]  = 0.0f;   // 512 entries, 1 each
        #pragma unroll
        for (int r = 0; r < 2; ++r) {       // 1024 entries, 2 each
            int idx = tid + r * 512;
            c2s[idx >> 5][idx & 31] = 0.0f;
        }
    }
    __syncthreads();

    v8f acc[2][2] = {};   // [f-half slot][k-tile]: 32 VGPRs of f32 accumulators

    v4f xv[4];            // in-flight x staging registers (16 VGPRs)
    v4f cv[8];            // in-flight c staging registers (32 VGPRs)

    // Issue all 12 global_load_b128 for one chunk back-to-back (no waits between).
    auto issue_loads = [&](int c) {
        const int tcb = (c >> 1) * 32;
        const int fh  = c & 1;
        #pragma unroll
        for (int i = 0; i < 4; ++i) {
            int e = tid + i * 512, f4 = e & 3, nt = e >> 2, n = nt >> 5, t = nt & 31;
            xv[i] = *reinterpret_cast<const v4f*>(
                xg + (size_t)(n0 + n) * 4096 + (size_t)(tcb + t) * 32 + fh * 16 + f4 * 4);
        }
        #pragma unroll
        for (int i = 0; i < 8; ++i) {
            int e = tid + i * 512, f4 = e & 3, kt = e >> 2, k = kt >> 5, t = kt & 31;
            cv[i] = *reinterpret_cast<const v4f*>(
                cg + (size_t)k * 4096 + (size_t)(tcb + t) * 32 + fh * 16 + f4 * 4);
        }
    };

    issue_loads(0);

    for (int c = 0; c < 8; ++c) {
        const int fh = c & 1;

        // ---- convert + store current chunk into LDS (waits on the loads here)
        #pragma unroll
        for (int i = 0; i < 4; ++i) {
            int e = tid + i * 512, f4 = e & 3, nt = e >> 2, n = nt >> 5, t = nt & 31;
            ldsX[f4 * 4 + 0][n][t] = (_Float16)xv[i].x;
            ldsX[f4 * 4 + 1][n][t] = (_Float16)xv[i].y;
            ldsX[f4 * 4 + 2][n][t] = (_Float16)xv[i].z;
            ldsX[f4 * 4 + 3][n][t] = (_Float16)xv[i].w;
        }
        #pragma unroll
        for (int i = 0; i < 8; ++i) {
            int e = tid + i * 512, f4 = e & 3, kt = e >> 2, k = kt >> 5, t = kt & 31;
            ldsC[f4 * 4 + 0][k][t] = (_Float16)cv[i].x;
            ldsC[f4 * 4 + 1][k][t] = (_Float16)cv[i].y;
            ldsC[f4 * 4 + 2][k][t] = (_Float16)cv[i].z;
            ldsC[f4 * 4 + 3][k][t] = (_Float16)cv[i].w;
        }

        // ---- kick off next chunk's global loads (overlap with compute below)
        if (c < 7) issue_loads(c + 1);
        __syncthreads();

        // ---- x2 partial: first 8 waves, thread owns (n = tid>>4, f' = tid&15)
        if (tid < 256) {
            int n = tid >> 4, fp = tid & 15;
            const _Float16* xr = &ldsX[fp][n][0];
            float s = 0.0f;
            #pragma unroll
            for (int t = 0; t < 32; ++t) { float h = (float)xr[t]; s = fmaf(h, h, s); }
            x2s[n][fh * 16 + fp] += s;
        }
        // ---- c2 partial: thread owns 1 (k, f') pair
        {
            int k = tid >> 4, fp = tid & 15;
            const _Float16* cr = &ldsC[fp][k][0];
            float s = 0.0f;
            #pragma unroll
            for (int t = 0; t < 32; ++t) { float h = (float)cr[t]; s = fmaf(h, h, s); }
            c2s[k][fh * 16 + fp] += s;
        }

        // ---- WMMA: wave w handles feature fwl = w of this f-half (EXEC all-1s)
        {
            const int fwl = w;
            // A fragment: 16(M=n) x 32(K=t) f16, K = g*16 + i (contiguous, same as B)
            union { v16h v; unsigned int u[8]; } a;
            const unsigned int* xrow =
                reinterpret_cast<const unsigned int*>(&ldsX[fwl][m][0]);
            #pragma unroll
            for (int p = 0; p < 8; ++p) {
                int d = g * 8 + p; // dword idx in t-row
                a.u[p] = xrow[d];
            }
            #pragma unroll
            for (int kt = 0; kt < 2; ++kt) {
                // B fragment: 32(K=t) x 16(N=k); lane holds col k = kt*16+m,
                // K = g*16 + i  -> contiguous 32B in LDS
                union { v16h v; v4u q[2]; } b;
                const v4u* cp =
                    reinterpret_cast<const v4u*>(&ldsC[fwl][kt * 16 + m][g * 16]);
                b.q[0] = cp[0];
                b.q[1] = cp[1];
                acc[fh][kt] = __builtin_amdgcn_wmma_f32_16x16x32_f16(
                    false, a.v, false, b.v, (short)0, acc[fh][kt], false, false);
            }
        }
        __syncthreads();
    }

    // ---- epilogue: ed = sqrt(max(x2 + c2 - 2*xc, 0)), sum over this wave's
    // 2 features, then ds_add_f32 reduce across waves into dist[n][k].
    float edsum[2][8] = {};
    #pragma unroll
    for (int s = 0; s < 2; ++s) {
        int f = s * 16 + w;
        #pragma unroll
        for (int kt = 0; kt < 2; ++kt) {
            int k = kt * 16 + m;
            #pragma unroll
            for (int r = 0; r < 8; ++r) {
                int nrow = r + 8 * g;                  // D layout: M = r + 8*g, N = m
                float val = acc[s][kt][r];
                float sq  = x2s[nrow][f] + c2s[k][f] - 2.0f * val;
                edsum[kt][r] += sqrtf(fmaxf(sq, 0.0f));
            }
        }
    }
    #pragma unroll
    for (int kt = 0; kt < 2; ++kt)
        #pragma unroll
        for (int r = 0; r < 8; ++r)
            atomicAdd(&dist[r + 8 * g][kt * 16 + m], edsum[kt][r]);
    __syncthreads();

    // ---- Student-t soft assignment (ALPHA=1): q = 1/(1+d^2), normalized over k.
    // wave w handles row n = w; lane = k.
    {
        int n = w;
        float d  = dist[n][L];
        float qv = 1.0f / (1.0f + d * d);
        float qs = qv;
        #pragma unroll
        for (int off = 16; off >= 1; off >>= 1)
            qs += __shfl_xor(qs, off, 32);
        outg[(size_t)(n0 + n) * 32 + L] = qv / qs;
    }
}

extern "C" void kernel_launch(void* const* d_in, const int* in_sizes, int n_in,
                              void* d_out, int out_size, void* d_ws, size_t ws_size,
                              hipStream_t stream) {
    const float* x   = (const float*)d_in[0];   // (N,128,32) f32
    const float* c   = (const float*)d_in[1];   // (32,128,32) f32
    float*       out = (float*)d_out;           // (N,32) f32

    int N = in_sizes[0] / (128 * 32);
    int nblocks = N / 16;                       // 128 for N=2048
    tscluster_wmma_kernel<<<dim3(nblocks), dim3(512), 0, stream>>>(x, c, out);
}
